// GraphClassifier_66881230733786
// MI455X (gfx1250) — matrix operand order repaired and checked by reading, not compiled, not run on hardware
//
#include <hip/hip_runtime.h>

typedef __attribute__((ext_vector_type(16))) _Float16 v16h;
typedef __attribute__((ext_vector_type(8)))  _Float16 v8h;
typedef __attribute__((ext_vector_type(8)))  float    v8f;
typedef __attribute__((ext_vector_type(4)))  float    v4f;

static constexpr int kN = 50000;     // nodes
static constexpr int kE = 1250000;   // edges
static constexpr int kG = 256;       // graphs
static constexpr int kH = 64;        // feature dim (IN == H == 64)
static constexpr int kTiles = kN / 16;  // 3125 exactly
static constexpr int kWFrag = 4096;     // halves per packed 64x64 weight matrix

// ---------------------------------------------------------------------------
// Float atomic max via integer punning (works with amax initialized to -inf)
// ---------------------------------------------------------------------------
__device__ __forceinline__ void atomicMaxF(float* addr, float val) {
    if (val >= 0.0f)
        atomicMax((int*)addr, __float_as_int(val));
    else
        atomicMin((unsigned int*)addr, __float_as_uint(val));
}

// ---------------------------------------------------------------------------
// One-shot: pack eight 64x64 f32 weight matrices into f16 WMMA B-fragment
// lane order: out[mat][ (ks*4+ct)*32 + lane ][h]  =  W[ks*32 + (h&8)*2 + (lane>>4)*8 + (h&7)]
//                                                     [ct*16 + (lane&15)]
// so each lane's 16 halves are one contiguous 32-byte run.
// ---------------------------------------------------------------------------
__global__ __launch_bounds__(256)
void weight_pack(const float* __restrict__ W0, const float* __restrict__ W1,
                 const float* __restrict__ W2, const float* __restrict__ W3,
                 const float* __restrict__ W4, const float* __restrict__ W5,
                 const float* __restrict__ W6, const float* __restrict__ W7,
                 _Float16* __restrict__ out)
{
    const int t = blockIdx.x * blockDim.x + threadIdx.x;   // 0 .. 8*4096-1
    const int mat = t >> 12;
    const int idx = t & (kWFrag - 1);
    const int h    = idx & 15;
    const int lane = (idx >> 4) & 31;
    const int ct   = (idx >> 9) & 3;
    const int ks   = idx >> 11;
    const int K = ks * 32 + (h & 8) * 2 + (lane >> 4) * 8 + (h & 7);
    const int n = ct * 16 + (lane & 15);
    const float* W;
    switch (mat) {
        case 0: W = W0; break; case 1: W = W1; break;
        case 2: W = W2; break; case 3: W = W3; break;
        case 4: W = W4; break; case 5: W = W5; break;
        case 6: W = W6; break; default: W = W7; break;
    }
    out[t] = (_Float16)W[K * kH + n];
}

// ---------------------------------------------------------------------------
// One-shot: cast layer-1 input activations f32 -> f16 (row-major unchanged)
// ---------------------------------------------------------------------------
__global__ __launch_bounds__(256)
void cast_f16(const float* __restrict__ X, _Float16* __restrict__ Xh)
{
    const int t = blockIdx.x * blockDim.x + threadIdx.x;
    if (t < kN * kH) Xh[t] = (_Float16)X[t];
}

// ---------------------------------------------------------------------------
// Fused Q/K/V/S GEMM:  O = X @ W + b  for four weight sets.
// One wave -> one 16x64 tile of one output matrix, via v_wmma_f32_16x16x32_f16.
// Inner loop: A = 2x b128 loads from f16 X, B = 1x 32B load from packed f16 W.
// Block = 256 threads = 8 waves = 2 row tiles x 4 matrices.
// ---------------------------------------------------------------------------
__global__ __launch_bounds__(256)
void gemm_qkvs(const _Float16* __restrict__ Xh,
               const _Float16* __restrict__ Wpack,   // 4 packed matrices
               const float* __restrict__ bq, const float* __restrict__ bk,
               const float* __restrict__ bv, const float* __restrict__ bs,
               float* __restrict__ Q, float* __restrict__ Kout,
               float* __restrict__ V, float* __restrict__ S)
{
    const int wave = threadIdx.x >> 5;
    const int lane = threadIdx.x & 31;
    const int mat  = wave & 3;
    const int tile = blockIdx.x * 2 + (wave >> 2);
    if (tile >= kTiles) return;              // wave-uniform: EXEC stays all-ones

    const float* b; float* O;
    switch (mat) {
        case 0:  b = bq; O = Q;    break;
        case 1:  b = bk; O = Kout; break;
        case 2:  b = bv; O = V;    break;
        default: b = bs; O = S;    break;
    }
    const _Float16* wf = Wpack + (size_t)mat * kWFrag;

    const int m    = lane & 15;
    const int hi   = lane >> 4;     // lane half selects K sub-block
    const int row0 = tile * 16;

    v8f acc[4] = {};                // four 16x16 f32 accumulators (N tiles)

    #pragma unroll
    for (int ks = 0; ks < 2; ++ks) {
        const int kk = ks * 32;
        // ---- A fragment: halves 0..7 = K kk+hi*8.. , halves 8..15 = +16
        const _Float16* xr = Xh + (size_t)(row0 + m) * kH + kk + hi * 8;
        const v8h alo = *(const v8h*)(xr);
        const v8h ahi = *(const v8h*)(xr + 16);
        const v16h a = __builtin_shufflevector(alo, ahi,
            0, 1, 2, 3, 4, 5, 6, 7, 8, 9, 10, 11, 12, 13, 14, 15);
        // ---- B fragments: pre-swizzled, contiguous 16 halves per lane
        #pragma unroll
        for (int ct = 0; ct < 4; ++ct) {
            const v16h bf =
                *(const v16h*)(wf + (((ks * 4 + ct) * 32 + lane) << 4));
            acc[ct] = __builtin_amdgcn_wmma_f32_16x16x32_f16(
                false, a, false, bf, (short)0, acc[ct], false, false);
        }
    }

    // ---- store D (C/D layout: VGPR r -> M = r + hi*8, N = ct*16 + m)
    #pragma unroll
    for (int ct = 0; ct < 4; ++ct) {
        const int col  = ct * 16 + m;
        const float bb = b[col];
        #pragma unroll
        for (int r = 0; r < 8; ++r) {
            const int row = row0 + r + hi * 8;
            O[(size_t)row * kH + col] = acc[ct][r] + bb;
        }
    }
}

// ---------------------------------------------------------------------------
// Per-layer segment-state init
// ---------------------------------------------------------------------------
__global__ __launch_bounds__(256)
void layer_init(float* __restrict__ agg, float* __restrict__ amax,
                float* __restrict__ denom)
{
    const int t = blockIdx.x * blockDim.x + threadIdx.x;
    if (t < kN * kH) agg[t] = 0.0f;
    if (t < kN) { amax[t] = -__builtin_huge_valf(); denom[t] = 0.0f; }
}

// ---------------------------------------------------------------------------
// Pass A: alpha_e = <q[dst], k[src] + w_e * We> / 8 ;  segment max into amax
// ---------------------------------------------------------------------------
__global__ __launch_bounds__(256)
void edge_alpha(const float* __restrict__ Q, const float* __restrict__ K,
                const int* __restrict__ src, const int* __restrict__ dst,
                const float* __restrict__ ew, const float* __restrict__ We,
                float* __restrict__ alpha, float* __restrict__ amax)
{
    const int e = blockIdx.x * blockDim.x + threadIdx.x;
    if (e >= kE) return;
    const int s = src[e], d = dst[e];
    const float w = ew[e];
    const float* q = Q + (size_t)d * kH;
    const float* k = K + (size_t)s * kH;
    float sum = 0.0f;
    #pragma unroll
    for (int c = 0; c < kH; c += 4) {
        const v4f qv = *(const v4f*)(q + c);
        const v4f kv = *(const v4f*)(k + c);
        const v4f evv = *(const v4f*)(We + c);
        sum += qv[0] * (kv[0] + w * evv[0]) + qv[1] * (kv[1] + w * evv[1])
             + qv[2] * (kv[2] + w * evv[2]) + qv[3] * (kv[3] + w * evv[3]);
    }
    const float a = sum * 0.125f;   // 1/sqrt(64)
    alpha[e] = a;
    atomicMaxF(&amax[d], a);
}

// ---------------------------------------------------------------------------
// Pass B: ex_e = exp(alpha_e - amax[dst]) ; segment sum into denom
// ---------------------------------------------------------------------------
__global__ __launch_bounds__(256)
void edge_exp(const int* __restrict__ dst, const float* __restrict__ alpha,
              const float* __restrict__ amax, float* __restrict__ ex,
              float* __restrict__ denom)
{
    const int e = blockIdx.x * blockDim.x + threadIdx.x;
    if (e >= kE) return;
    const int d = dst[e];
    float m = amax[d];
    if (m < -1e37f) m = 0.0f;       // isfinite() replacement for -inf init
    const float v = expf(alpha[e] - m);
    ex[e] = v;
    atomicAdd(&denom[d], v);
}

// ---------------------------------------------------------------------------
// Pass C: agg[dst] += (ex/denom) * (v[src] + w_e * We)   (64 lanes per edge)
// ---------------------------------------------------------------------------
__global__ __launch_bounds__(256)
void edge_agg(const int* __restrict__ src, const int* __restrict__ dst,
              const float* __restrict__ ew, const float* __restrict__ We,
              const float* __restrict__ V, const float* __restrict__ ex,
              const float* __restrict__ denom, float* __restrict__ agg)
{
    const long long t = (long long)blockIdx.x * blockDim.x + threadIdx.x;
    if (t >= (long long)kE * kH) return;
    const int e = (int)(t >> 6);
    const int c = (int)(t & 63);
    const int s = src[e], d = dst[e];
    float dn = denom[d];
    dn = dn > 1e-16f ? dn : 1e-16f;
    const float w = ex[e] / dn;
    const float val = w * (V[(size_t)s * kH + c] + ew[e] * We[c]);
    atomicAdd(&agg[(size_t)d * kH + c], val);
}

// ---------------------------------------------------------------------------
// H = relu(agg + S)  -> f32 (for pooling / inspection) and f16 (next GEMM)
// ---------------------------------------------------------------------------
__global__ __launch_bounds__(256)
void finalize_relu(const float* __restrict__ agg, const float* __restrict__ S,
                   float* __restrict__ H, _Float16* __restrict__ Hh)
{
    const int t = blockIdx.x * blockDim.x + threadIdx.x;
    if (t >= kN * kH) return;
    float v = agg[t] + S[t];
    v = v > 0.0f ? v : 0.0f;
    H[t]  = v;
    Hh[t] = (_Float16)v;
}

// ---------------------------------------------------------------------------
// Mean pool per graph + final linear head
// ---------------------------------------------------------------------------
__global__ __launch_bounds__(256)
void pool_init(float* __restrict__ pooled, float* __restrict__ cnt)
{
    const int t = blockIdx.x * blockDim.x + threadIdx.x;
    if (t < kG * kH) pooled[t] = 0.0f;
    if (t < kG) cnt[t] = 0.0f;
}

__global__ __launch_bounds__(256)
void pool_sum(const float* __restrict__ H, const int* __restrict__ batch,
              float* __restrict__ pooled, float* __restrict__ cnt)
{
    const int t = blockIdx.x * blockDim.x + threadIdx.x;
    if (t >= kN * kH) return;
    const int n = t >> 6, c = t & 63;
    const int g = batch[n];
    atomicAdd(&pooled[(size_t)g * kH + c], H[t]);
    if (c == 0) atomicAdd(&cnt[g], 1.0f);
}

__global__ __launch_bounds__(256)
void head(const float* __restrict__ pooled, const float* __restrict__ cnt,
          const float* __restrict__ Wl, const float* __restrict__ bl,
          float* __restrict__ out)
{
    const int g = threadIdx.x;
    if (g >= kG) return;
    float c = cnt[g];
    c = c > 1.0f ? c : 1.0f;
    const float inv = 1.0f / c;
    float o0 = bl[0], o1 = bl[1];
    #pragma unroll
    for (int k = 0; k < kH; ++k) {
        const float p = pooled[(size_t)g * kH + k] * inv;
        o0 += p * Wl[k * 2 + 0];
        o1 += p * Wl[k * 2 + 1];
    }
    out[g * 2 + 0] = o0;
    out[g * 2 + 1] = o1;
}

// ---------------------------------------------------------------------------
// Host launcher
// ---------------------------------------------------------------------------
extern "C" void kernel_launch(void* const* d_in, const int* in_sizes, int n_in,
                              void* d_out, int out_size, void* d_ws, size_t ws_size,
                              hipStream_t stream)
{
    (void)in_sizes; (void)n_in; (void)out_size; (void)ws_size;

    const float* x     = (const float*)d_in[0];
    const int*   ei    = (const int*)d_in[1];
    const int*   src   = ei;
    const int*   dst   = ei + kE;
    const float* ew    = (const float*)d_in[2];
    const int*   batch = (const int*)d_in[3];

    const float* Wq1 = (const float*)d_in[4];  const float* bq1 = (const float*)d_in[5];
    const float* Wk1 = (const float*)d_in[6];  const float* bk1 = (const float*)d_in[7];
    const float* Wv1 = (const float*)d_in[8];  const float* bv1 = (const float*)d_in[9];
    const float* We1 = (const float*)d_in[10];
    const float* Ws1 = (const float*)d_in[11]; const float* bs1 = (const float*)d_in[12];
    const float* Wq2 = (const float*)d_in[13]; const float* bq2 = (const float*)d_in[14];
    const float* Wk2 = (const float*)d_in[15]; const float* bk2 = (const float*)d_in[16];
    const float* Wv2 = (const float*)d_in[17]; const float* bv2 = (const float*)d_in[18];
    const float* We2 = (const float*)d_in[19];
    const float* Ws2 = (const float*)d_in[20]; const float* bs2 = (const float*)d_in[21];
    const float* Wl  = (const float*)d_in[22]; const float* bl  = (const float*)d_in[23];

    // Workspace carve-up (float units, 256B-aligned chunks). Total ~ 95 MB.
    float* ws = (float*)d_ws;
    size_t off = 0;
    auto alloc = [&](size_t nelem) {
        float* p = ws + off;
        off += (nelem + 63) & ~(size_t)63;
        return p;
    };
    float* Q      = alloc((size_t)kN * kH);
    float* Kb     = alloc((size_t)kN * kH);
    float* V      = alloc((size_t)kN * kH);
    float* S      = alloc((size_t)kN * kH);
    float* Hbuf   = alloc((size_t)kN * kH);
    float* agg    = alloc((size_t)kN * kH);
    float* alphaB = alloc((size_t)kE);
    float* exB    = alloc((size_t)kE);
    float* amax   = alloc((size_t)kN);
    float* denom  = alloc((size_t)kN);
    float* pooled = alloc((size_t)kG * kH);
    float* cnt    = alloc((size_t)kG);
    _Float16* Xh    = (_Float16*)alloc((size_t)kN * kH / 2);   // f16 activations
    _Float16* Wpack = (_Float16*)alloc((size_t)8 * kWFrag / 2); // packed weights

    const int nhBlocks    = (kN * kH) / 256;         // 12,500
    const int gemmBlocks  = (kTiles + 1) / 2;        // 1,563
    const int edgeBlocks  = (kE + 255) / 256;        // 4,883
    const int eaggBlocks  = (int)(((long long)kE * kH) / 256);  // 312,500

    // One-shot prep: pack all 8 weight matrices, cast layer-1 activations
    weight_pack<<<(8 * kWFrag) / 256, 256, 0, stream>>>(
        Wq1, Wk1, Wv1, Ws1, Wq2, Wk2, Wv2, Ws2, Wpack);
    cast_f16<<<nhBlocks, 256, 0, stream>>>(x, Xh);

    auto run_layer = [&](const _Float16* Wp,
                         const float* bq, const float* bk,
                         const float* bv, const float* We,
                         const float* bsk) {
        layer_init   <<<nhBlocks,   256, 0, stream>>>(agg, amax, denom);
        gemm_qkvs    <<<gemmBlocks, 256, 0, stream>>>(Xh, Wp, bq, bk, bv, bsk,
                                                      Q, Kb, V, S);
        edge_alpha   <<<edgeBlocks, 256, 0, stream>>>(Q, Kb, src, dst, ew, We,
                                                      alphaB, amax);
        edge_exp     <<<edgeBlocks, 256, 0, stream>>>(dst, alphaB, amax, exB, denom);
        edge_agg     <<<eaggBlocks, 256, 0, stream>>>(src, dst, ew, We, V, exB,
                                                      denom, agg);
        finalize_relu<<<nhBlocks,   256, 0, stream>>>(agg, S, Hbuf, Xh);
    };

    run_layer(Wpack,              bq1, bk1, bv1, We1, bs1);
    run_layer(Wpack + 4 * kWFrag, bq2, bk2, bv2, We2, bs2);

    pool_init<<<(kG * kH + 255) / 256, 256, 0, stream>>>(pooled, cnt);
    pool_sum <<<nhBlocks,              256, 0, stream>>>(Hbuf, batch, pooled, cnt);
    head     <<<1,                     256, 0, stream>>>(pooled, cnt, Wl, bl,
                                                         (float*)d_out);
}
